// CosineRouter_8770323218989
// MI455X (gfx1250) — matrix operand order repaired and checked by reading, not compiled, not run on hardware
//
#include <hip/hip_runtime.h>
#include <hip/hip_bf16.h>

// ---------------------------------------------------------------------------
// CosineRouter for MI455X (gfx1250, wave32, WMMA)
//   x:(64,2048,1024) f32, W:(256,1024) f32, b:(256) f32, centers:(512,256) f32
//   out: topk_probs f32 (131072,2)  ++  topk_idx i32 (131072,2)
// Strategy: bf16 WMMA (f32 accum) -> memory bound (~512MB of x @ 23.3TB/s).
// d_ws layout: [0, 512KB) W in bf16 ; [512KB, 768KB) normalized centers bf16.
// ---------------------------------------------------------------------------

typedef __attribute__((ext_vector_type(16))) __bf16 v16bf;
typedef __attribute__((ext_vector_type(8)))  float  v8f;

#define N_TOKENS   131072          // 64*2048
#define IN_DIM     1024
#define E_DIM      256
#define N_EXPERTS  512
#define TILE_M     16
#define A1_LD      1040            // bf16 elems, multiple of 8 -> 16B aligned rows
#define A2_LD      264
#define EPS        1e-12f

union FragU { v16bf v; uint4 q[2]; };

// A-matrix 16xK bf16 fragment: lane row = l&15, kOff = (l>>4)*16,
// VGPR0..3 = 8 contiguous elems at +0, VGPR4..7 = 8 contiguous elems at +8.
__device__ __forceinline__ v16bf load_a_frag(const __bf16* p) {
  FragU f;
  f.q[0] = *reinterpret_cast<const uint4*>(p);
  f.q[1] = *reinterpret_cast<const uint4*>(p + 8);
  return f.v;
}

// B-matrix Kx16 bf16 fragment, source stored as [col][k] row-major:
// lane col = l&15, kOff = (l>>4)*16, 16 contiguous elems (halves at +0, +8).
__device__ __forceinline__ v16bf load_b_frag(const __bf16* p) {
  FragU f;
  f.q[0] = *reinterpret_cast<const uint4*>(p);
  f.q[1] = *reinterpret_cast<const uint4*>(p + 8);
  return f.v;
}

__device__ __forceinline__ v8f wmma_bf16(v16bf a, v16bf b, v8f c) {
  return __builtin_amdgcn_wmma_f32_16x16x32_bf16(
      /*neg_a=*/false, a, /*neg_b=*/false, b,
      /*c_mod=*/(short)0, c, /*reuse_a=*/false, /*reuse_b=*/false);
}

// ----------------------------- prep kernels --------------------------------

__global__ __launch_bounds__(256) void convert_w_kernel(const float* __restrict__ W,
                                                        __bf16* __restrict__ Wbf) {
  // 256 blocks * 256 threads * 4 elems = 262144
  int i = (blockIdx.x * 256 + threadIdx.x) * 4;
  float4 f = *reinterpret_cast<const float4*>(W + i);
  Wbf[i + 0] = (__bf16)f.x;
  Wbf[i + 1] = (__bf16)f.y;
  Wbf[i + 2] = (__bf16)f.z;
  Wbf[i + 3] = (__bf16)f.w;
}

__global__ __launch_bounds__(256) void norm_centers_kernel(const float* __restrict__ centers,
                                                           __bf16* __restrict__ cn) {
  __shared__ float red[256];
  int r = blockIdx.x;                 // one expert row per block
  int t = threadIdx.x;
  float v = centers[r * E_DIM + t];
  red[t] = v * v;
  __syncthreads();
  for (int s = 128; s > 0; s >>= 1) {
    if (t < s) red[t] += red[t + s];
    __syncthreads();
  }
  float inv = 1.0f / fmaxf(sqrtf(red[0]), EPS);
  cn[r * E_DIM + t] = (__bf16)(v * inv);
}

// ------------------------------ main kernel --------------------------------

union SmemU {
  __bf16 a1[TILE_M][A1_LD];          // bf16 x tile (GEMM1 A)      33280 B
  float  sim[TILE_M][N_EXPERTS + 1]; // cosine sims (reused space) 32832 B
};

__global__ __launch_bounds__(256) void router_kernel(
    const float* __restrict__ x, const __bf16* __restrict__ Wbf,
    const float* __restrict__ bias, const __bf16* __restrict__ cn,
    float* __restrict__ outProbs, int* __restrict__ outIdx) {
  __shared__ SmemU u;
  __shared__ float  xproj[TILE_M][E_DIM + 1];   // 16448 B
  __shared__ __bf16 a2[TILE_M][A2_LD];          //  8448 B
  __shared__ float  partial[TILE_M][16];        //  1024 B
  __shared__ float  invNorm[TILE_M];
  __shared__ float  cv1[TILE_M][16], cv2[TILE_M][16];
  __shared__ int    ci1[TILE_M][16], ci2[TILE_M][16];

  const int tid  = threadIdx.x;
  const int wv   = tid >> 5;           // wave 0..7
  const int lane = tid & 31;
  const int row  = lane & 15;          // A row / B col within a 16-wide tile
  const int aOff = (lane >> 4) * 16;   // A fragment k offset
  const int bOff = (lane >> 4) * 16;   // B fragment k offset
  const int mb   = (lane >> 4) * 8;    // C/D: M base for this lane

  // ---- stage 1: load 16x1024 f32 x tile, convert to bf16 in LDS ----
  {
    const size_t base = (size_t)blockIdx.x * (TILE_M * IN_DIM);
    const float4* x4 = reinterpret_cast<const float4*>(x + base);
    #pragma unroll
    for (int i = 0; i < 16; ++i) {
      int idx4 = tid + i * 256;                  // 4096 float4s total
      float4 f = x4[idx4];
      int flat = idx4 * 4;
      int m = flat >> 10;
      int k = flat & (IN_DIM - 1);
      u.a1[m][k + 0] = (__bf16)f.x;
      u.a1[m][k + 1] = (__bf16)f.y;
      u.a1[m][k + 2] = (__bf16)f.z;
      u.a1[m][k + 3] = (__bf16)f.w;
    }
  }
  __syncthreads();

  // ---- stage 2: GEMM1  x_proj[16,256] = x[16,1024] * W^T + b ----
  {
    const int e0 = wv * 32;                      // 2 expert tiles per wave
    const __bf16* Wr0 = Wbf + (size_t)(e0 + row) * IN_DIM;
    const __bf16* Wr1 = Wbf + (size_t)(e0 + 16 + row) * IN_DIM;
    v8f acc0 = {}; v8f acc1 = {};
    #pragma unroll 4
    for (int ks = 0; ks < IN_DIM; ks += 32) {
      v16bf a  = load_a_frag(&u.a1[row][ks + aOff]);
      v16bf b0 = load_b_frag(Wr0 + ks + bOff);
      acc0 = wmma_bf16(a, b0, acc0);
      v16bf b1 = load_b_frag(Wr1 + ks + bOff);
      acc1 = wmma_bf16(a, b1, acc1);
    }
    float bi0 = bias[e0 + row];
    float bi1 = bias[e0 + 16 + row];
    #pragma unroll
    for (int v = 0; v < 8; ++v) {
      xproj[mb + v][e0 + row]      = acc0[v] + bi0;
      xproj[mb + v][e0 + 16 + row] = acc1[v] + bi1;
    }
  }
  __syncthreads();

  // ---- stage 3: row L2 norms over 256, renormalize into bf16 A2 ----
  {
    const int r = tid >> 4, seg = tid & 15;
    float s = 0.f;
    #pragma unroll
    for (int j = 0; j < 16; ++j) {
      float v = xproj[r][seg * 16 + j];
      s += v * v;
    }
    partial[r][seg] = s;
    __syncthreads();
    if (seg == 0) {
      float tot = 0.f;
      #pragma unroll
      for (int j = 0; j < 16; ++j) tot += partial[r][j];
      invNorm[r] = 1.0f / fmaxf(sqrtf(tot), EPS);
    }
    __syncthreads();
    float inv = invNorm[r];
    #pragma unroll
    for (int j = 0; j < 16; ++j)
      a2[r][seg * 16 + j] = (__bf16)(xproj[r][seg * 16 + j] * inv);
  }
  __syncthreads();

  // ---- stage 4: GEMM2  sim[16,512] = x_proj_n[16,256] * c_n^T ----
  {
    const int c0 = wv * 64;                      // 4 expert tiles per wave
    v8f s0 = {}, s1 = {}, s2 = {}, s3 = {};
    const __bf16* C0 = cn + (size_t)(c0 + row) * E_DIM;
    const __bf16* C1 = cn + (size_t)(c0 + 16 + row) * E_DIM;
    const __bf16* C2 = cn + (size_t)(c0 + 32 + row) * E_DIM;
    const __bf16* C3 = cn + (size_t)(c0 + 48 + row) * E_DIM;
    #pragma unroll
    for (int ks = 0; ks < E_DIM; ks += 32) {
      v16bf a = load_a_frag(&a2[row][ks + aOff]);
      s0 = wmma_bf16(a, load_b_frag(C0 + ks + bOff), s0);
      s1 = wmma_bf16(a, load_b_frag(C1 + ks + bOff), s1);
      s2 = wmma_bf16(a, load_b_frag(C2 + ks + bOff), s2);
      s3 = wmma_bf16(a, load_b_frag(C3 + ks + bOff), s3);
    }
    #pragma unroll
    for (int v = 0; v < 8; ++v) {
      u.sim[mb + v][c0 + row]      = s0[v];
      u.sim[mb + v][c0 + 16 + row] = s1[v];
      u.sim[mb + v][c0 + 32 + row] = s2[v];
      u.sim[mb + v][c0 + 48 + row] = s3[v];
    }
  }
  __syncthreads();

  // ---- stage 5: top-2 over 512 + softmax (jax tie rule: lower index wins) ----
  {
    const int r = tid >> 4, seg = tid & 15;
    float v1 = -INFINITY, v2 = -INFINITY; int i1 = 0, i2 = 0;
    const int base = seg * 32;
    for (int j = 0; j < 32; ++j) {
      float v = u.sim[r][base + j];
      if (v > v1)      { v2 = v1; i2 = i1; v1 = v; i1 = base + j; }
      else if (v > v2) { v2 = v; i2 = base + j; }
    }
    cv1[r][seg] = v1; ci1[r][seg] = i1;
    cv2[r][seg] = v2; ci2[r][seg] = i2;
    __syncthreads();
    if (tid < TILE_M) {
      float b1 = -INFINITY, b2 = -INFINITY; int j1 = 0, j2 = 0;
      for (int s = 0; s < 16; ++s) {
        float a = cv1[tid][s]; int ia = ci1[tid][s];
        if (a > b1)      { b2 = b1; j2 = j1; b1 = a; j1 = ia; }
        else if (a > b2) { b2 = a; j2 = ia; }
        float c = cv2[tid][s]; int ic = ci2[tid][s];
        if (c > b1)      { b2 = b1; j2 = j1; b1 = c; j1 = ic; }
        else if (c > b2) { b2 = c; j2 = ic; }
      }
      float p1 = 1.0f / (1.0f + __expf(b2 - b1));   // softmax over [b1,b2]
      int token = blockIdx.x * TILE_M + tid;
      outProbs[token * 2 + 0] = p1;
      outProbs[token * 2 + 1] = 1.0f - p1;
      outIdx[token * 2 + 0] = j1;
      outIdx[token * 2 + 1] = j2;
    }
  }
}

// ------------------------------- launcher ----------------------------------

extern "C" void kernel_launch(void* const* d_in, const int* in_sizes, int n_in,
                              void* d_out, int out_size, void* d_ws, size_t ws_size,
                              hipStream_t stream) {
  const float* x       = (const float*)d_in[0];
  const float* W       = (const float*)d_in[1];
  const float* b       = (const float*)d_in[2];
  const float* centers = (const float*)d_in[3];
  // d_in[4] = top_k (fixed at 2 for this problem shape)

  float* outProbs = (float*)d_out;
  int*   outIdx   = (int*)((float*)d_out + (size_t)2 * N_TOKENS);

  __bf16* Wbf = (__bf16*)d_ws;                                          // 512 KB
  __bf16* cnb = (__bf16*)((char*)d_ws + (size_t)E_DIM * IN_DIM * 2);    // 256 KB

  convert_w_kernel<<<256, 256, 0, stream>>>(W, Wbf);
  norm_centers_kernel<<<N_EXPERTS, 256, 0, stream>>>(centers, cnb);
  router_kernel<<<N_TOKENS / TILE_M, 256, 0, stream>>>(x, Wbf, b, cnb,
                                                       outProbs, outIdx);
}
